// MultiHeadAttentionLlama2_22617297780794
// MI455X (gfx1250) — compile-verified, matrix operand order checked by reading, not run on hardware
//
#include <hip/hip_runtime.h>
#include <hip/hip_bf16.h>

// ---------------------------------------------------------------------------
// MHA forward for MI455X (gfx1250): bf16 WMMA (16x16x32) with f32 accumulate.
//
// GEMMs (QKV + proj): 128x128 block tile, 8 waves (wave = 32x64 = 2x4 WMMA
// tiles), K staged through LDS with gfx1250 async global->LDS copies
// (global_load_async_to_lds_b128, ASYNCcnt), double buffered with the buffer
// index fully unrolled so LDS reads lower to ds_load_b128 (not flat).
//
// Attention: flash-style causal, per-wave 16 q-rows x 32-key blocks,
// online softmax with half-wave shuffles, P C-layout->A-layout transpose
// through a per-wave LDS bounce (s_wait_dscnt).
// ---------------------------------------------------------------------------

typedef __bf16 bf16;
typedef __attribute__((ext_vector_type(16))) __bf16 bf16x16;
typedef __attribute__((ext_vector_type(8)))  __bf16 bf16x8;
typedef __attribute__((ext_vector_type(4)))  __bf16 bf16x4;
typedef __attribute__((ext_vector_type(8)))  float  f32x8;

#define DEVINL __device__ __forceinline__

static constexpr int Bc = 4, Tc = 2048, Cc = 1024, Hc = 16, Dc = 64;
static constexpr int Mrows = Bc * Tc;          // 8192
static constexpr float kScale = 0.03125f;       // C^-0.5 = 1/32

DEVINL f32x8 f32x8_zero() {
  f32x8 z = {0.f,0.f,0.f,0.f,0.f,0.f,0.f,0.f};
  return z;
}

DEVINL bf16x16 frag_cat(bf16x8 lo, bf16x8 hi) {
  return __builtin_shufflevector(lo, hi, 0,1,2,3,4,5,6,7,8,9,10,11,12,13,14,15);
}

// p points at element k0 of this lane's row; hi = lane>>4.
// Lanes 0-15 hold k 0..7 and 16..23; lanes 16-31 hold k 8..15 and 24..31.
DEVINL bf16x16 load_frag(const bf16* p, int hi) {
  const bf16* q = p + hi * 8;
  bf16x8 a = *(const bf16x8*)(q);
  bf16x8 b = *(const bf16x8*)(q + 16);
  return frag_cat(a, b);
}

DEVINL f32x8 wmma_bf16(bf16x16 a, bf16x16 b, f32x8 c) {
  return __builtin_amdgcn_wmma_f32_16x16x32_bf16(
      /*neg_a=*/false, a, /*neg_b=*/false, b,
      /*c_mod=*/(short)0, c, /*reuse_a=*/false, /*reuse_b=*/false);
}

// LDS aperture maps flat addr[31:0] directly to the LDS byte offset
// (ISA 10.2 per-aperture calc), so the low 32 bits of a generic pointer to a
// __shared__ object are the LDS address the async-copy unit wants.
DEVINL unsigned lds_off_of(const void* p) {
  return (unsigned)(unsigned long long)p;
}

// One async 16-byte global->LDS copy per lane (GV mode, ASYNCcnt-tracked).
DEVINL void async_cp16(unsigned lds_off, const bf16* g) {
  unsigned long long ga = (unsigned long long)g;
  asm volatile("global_load_async_to_lds_b128 %0, %1, off"
               :: "v"(lds_off), "v"(ga) : "memory");
}

DEVINL void wait_async_le4() { asm volatile("s_wait_asynccnt 0x4" ::: "memory"); }
DEVINL void wait_async_le0() { asm volatile("s_wait_asynccnt 0x0" ::: "memory"); }

// ---------------------------------------------------------------------------
// fp32 -> bf16 conversion (vectorized: float4 in, 4x bf16 out)
// ---------------------------------------------------------------------------
__global__ void cvt_f32_bf16(const float* __restrict__ src,
                             bf16* __restrict__ dst, int n4) {
  int i = blockIdx.x * blockDim.x + threadIdx.x;
  if (i < n4) {
    float4 v = ((const float4*)src)[i];
    bf16x4 o;
    o[0] = (bf16)v.x; o[1] = (bf16)v.y; o[2] = (bf16)v.z; o[3] = (bf16)v.w;
    ((bf16x4*)dst)[i] = o;
  }
}

// One double-buffer phase: 2 A frags + 4 B frags from LDS, 8 WMMAs.
// Ab/Bb are passed as literal &As[0][0] / &As[1][0] so address-space
// inference keeps the loads on the ds path.
DEVINL void compute_step(const bf16* Ab, const bf16* Bb,
                         int aoff, int boff, int hi, f32x8 (&acc)[2][4]) {
  bf16x16 a[2], b[4];
#pragma unroll
  for (int i = 0; i < 2; ++i) a[i] = load_frag(Ab + aoff + i * 16 * 32, hi);
#pragma unroll
  for (int j = 0; j < 4; ++j) b[j] = load_frag(Bb + boff + j * 16 * 32, hi);
#pragma unroll
  for (int i = 0; i < 2; ++i)
#pragma unroll
    for (int j = 0; j < 4; ++j) acc[i][j] = wmma_bf16(a[i], b[j], acc[i][j]);
}

// ---------------------------------------------------------------------------
// Shared GEMM mainloop: Y[m,n] = sum_k X[m,k]*W[n,k], K=1024, block 128x128.
// A/B tiles (128x32 bf16 = 8KB each) staged to LDS via async copies, double
// buffered (buffer index compile-time constant at every call site).
// Each of the 256 threads copies 2x16B per tile per k-step.
// ---------------------------------------------------------------------------
DEVINL void gemm_mainloop(const bf16* __restrict__ X, const bf16* __restrict__ W,
                          int m0blk, int n0blk, int tid, f32x8 (&acc)[2][4]) {
  const int K = Cc;
  __shared__ bf16 As[2][128 * 32];
  __shared__ bf16 Bs[2][128 * 32];

  int lane = tid & 31;
  int lr   = lane & 15;
  int hi   = lane >> 4;
  int wid  = tid >> 5;
  int waveM = wid & 3, waveN = wid >> 2;

  // --- staging assignment: thread -> (row r, 16B-quarter qbase..qbase+1) ---
  int r     = tid >> 1;            // 0..127
  int qbase = (tid & 1) * 2;       // 0 or 2 (quarters of a 64B row)
  const bf16* gA0 = X + (long)(m0blk + r) * K + qbase * 8;
  const bf16* gB0 = W + (long)(n0blk + r) * K + qbase * 8;
  unsigned stOff = (unsigned)((r * 32 + qbase * 8) * (int)sizeof(bf16));
  unsigned uA0 = lds_off_of(&As[0][0]) + stOff;
  unsigned uA1 = lds_off_of(&As[1][0]) + stOff;
  unsigned uB0 = lds_off_of(&Bs[0][0]) + stOff;
  unsigned uB1 = lds_off_of(&Bs[1][0]) + stOff;

  // --- fragment read element offsets within a tile buffer ---
  int aoff = (waveM * 32 + lr) * 32;
  int boff = (waveN * 64 + lr) * 32;

  auto stage = [&](unsigned ua, unsigned ub, int k0) {
    const bf16* ga = gA0 + k0;
    const bf16* gb = gB0 + k0;
    async_cp16(ua,      ga);
    async_cp16(ua + 16, ga + 8);
    async_cp16(ub,      gb);
    async_cp16(ub + 16, gb + 8);
  };

#pragma unroll
  for (int i = 0; i < 2; ++i)
#pragma unroll
    for (int j = 0; j < 4; ++j) acc[i][j] = f32x8_zero();

  const int kSteps = K / 32;       // 32 (even)
  stage(uA0, uB0, 0);
#pragma unroll 1
  for (int s = 0; s < kSteps; s += 2) {
    // ---- phase 0: prefetch buf1, compute buf0 ----
    stage(uA1, uB1, (s + 1) * 32);       // s+1 <= 31 always
    wait_async_le4();                    // buf0's 4 copies retired
    __syncthreads();                     // RAW: all waves' tile landed
    compute_step(&As[0][0], &Bs[0][0], aoff, boff, hi, acc);
    __syncthreads();                     // WAR: buf0 free for next prefetch

    // ---- phase 1: prefetch buf0 (if more), compute buf1 ----
    if (s + 2 < kSteps) {
      stage(uA0, uB0, (s + 2) * 32);
      wait_async_le4();
    } else {
      wait_async_le0();
    }
    __syncthreads();
    compute_step(&As[1][0], &Bs[1][0], aoff, boff, hi, acc);
    __syncthreads();
  }
}

// ---------------------------------------------------------------------------
// QKV GEMM. vt_out==0: store [B,H,T,D] (Q,K). vt_out==1: store [B,H,D,T] (V^T).
// ---------------------------------------------------------------------------
__global__ __launch_bounds__(256)
void gemm_qkv(const bf16* __restrict__ X, const bf16* __restrict__ W,
              bf16* __restrict__ dst, int vt_out) {
  int tid  = threadIdx.x;
  int lane = tid & 31;
  int lr   = lane & 15;
  int hi   = lane >> 4;
  int wid  = tid >> 5;
  int waveM = wid & 3, waveN = wid >> 2;
  int m0blk = blockIdx.x * 128;
  int n0blk = blockIdx.y * 128;
  int m0 = m0blk + waveM * 32;
  int n0 = n0blk + waveN * 64;

  f32x8 acc[2][4];
  gemm_mainloop(X, W, m0blk, n0blk, tid, acc);

  // Epilogue. C layout: lane l, VGPR v -> row m = v + 8*(l>>4), col n = l&15.
#pragma unroll
  for (int i = 0; i < 2; ++i) {
#pragma unroll
    for (int j = 0; j < 4; ++j) {
      int gn = n0 + j * 16 + lr;      // output channel c
      int h  = gn >> 6;
      int d  = gn & 63;
      if (!vt_out) {
        // dst[((b*H+h)*T + t)*D + d],  m = b*T + t
#pragma unroll
        for (int v = 0; v < 8; ++v) {
          int m  = m0 + i * 16 + v + 8 * hi;
          int bb = m >> 11, t = m & (Tc - 1);
          dst[(((long)bb * Hc + h) * Tc + t) * Dc + d] = (bf16)acc[i][j][v];
        }
      } else {
        // V^T: dst[((b*H+h)*D + d)*T + t]; v runs over 8 consecutive t -> pack.
        int mb = m0 + i * 16 + 8 * hi;
        int bb = mb >> 11, t = mb & (Tc - 1);
        bf16x8 pk;
#pragma unroll
        for (int v = 0; v < 8; ++v) pk[v] = (bf16)acc[i][j][v];
        *(bf16x8*)&dst[(((long)bb * Hc + h) * Dc + d) * Tc + t] = pk;
      }
    }
  }
}

// ---------------------------------------------------------------------------
// Flash attention, causal. Q,K: [B,H,T,D]; Vt: [B,H,D,T]; O: [B,T,H,D] (bf16).
// 8 independent waves per block, each owns 16 query rows; key blocks of 32.
// Per step: 4 WMMAs for S, online softmax, P->LDS bounce, 4 WMMAs for O.
// ---------------------------------------------------------------------------
__global__ __launch_bounds__(256)
void attn_fwd(const bf16* __restrict__ Q, const bf16* __restrict__ Km,
              const bf16* __restrict__ Vt, bf16* __restrict__ O) {
  int tid  = threadIdx.x;
  int wid  = tid >> 5;
  int lane = tid & 31;
  int lr   = lane & 15;
  int hi   = lane >> 4;
  int h = blockIdx.y, b = blockIdx.z;
  long head = (long)b * Hc + h;
  const bf16* Qb = Q  + head * Tc * Dc;
  const bf16* Kb = Km + head * Tc * Dc;
  const bf16* Vb = Vt + head * Dc * Tc;
  int q0 = blockIdx.x * 128 + wid * 16;

  __shared__ bf16 pbuf[8][16 * 32];    // 1KB per-wave P-tile bounce buffer
  bf16* myp = &pbuf[wid][0];

  // Q fragments for this wave's 16 rows (d = 0..31, 32..63)
  const bf16* qrow = Qb + (long)(q0 + lr) * Dc;
  bf16x16 aq0 = load_frag(qrow + 0,  hi);
  bf16x16 aq1 = load_frag(qrow + 32, hi);

  f32x8 accO[4];
#pragma unroll
  for (int t4 = 0; t4 < 4; ++t4) accO[t4] = f32x8_zero();
  float mrun[8], lrun[8];
#pragma unroll
  for (int v = 0; v < 8; ++v) { mrun[v] = -__builtin_inff(); lrun[v] = 0.f; }

  int kend = q0 + 16;                   // keys needed: 0 .. q0+15
  for (int kb = 0; kb < kend; kb += 32) {
    // ---- S = Q K^T for 16x32 tile (2 n-subtiles x 2 d-chunks) ----
    f32x8 s[2];
#pragma unroll
    for (int j = 0; j < 2; ++j) {
      const bf16* krow = Kb + (long)(kb + j * 16 + lr) * Dc;
      bf16x16 b0 = load_frag(krow + 0,  hi);
      bf16x16 b1 = load_frag(krow + 32, hi);
      f32x8 t = wmma_bf16(aq0, b0, f32x8_zero());
      s[j]    = wmma_bf16(aq1, b1, t);
    }

    // ---- scale + causal mask + online softmax (per row v) ----
#pragma unroll
    for (int v = 0; v < 8; ++v) {
      int row = q0 + v + 8 * hi;
      int c0 = kb + lr, c1 = kb + 16 + lr;
      float s0 = (c0 <= row) ? s[0][v] * kScale : -__builtin_inff();
      float s1 = (c1 <= row) ? s[1][v] * kScale : -__builtin_inff();
      float mx = fmaxf(s0, s1);
#pragma unroll
      for (int off = 8; off; off >>= 1) mx = fmaxf(mx, __shfl_xor(mx, off, 32));
      float mnew = fmaxf(mrun[v], mx);
      float corr = __expf(mrun[v] - mnew);
      float p0 = __expf(s0 - mnew);
      float p1 = __expf(s1 - mnew);
      float rs = p0 + p1;
#pragma unroll
      for (int off = 8; off; off >>= 1) rs += __shfl_xor(rs, off, 32);
      lrun[v] = lrun[v] * corr + rs;
      mrun[v] = mnew;
#pragma unroll
      for (int t4 = 0; t4 < 4; ++t4) accO[t4][v] *= corr;
      // P tile -> LDS in [row][key] order (C-layout scatter)
      myp[(v + 8 * hi) * 32 + lr]      = (bf16)p0;
      myp[(v + 8 * hi) * 32 + 16 + lr] = (bf16)p1;
    }
    asm volatile("s_wait_dscnt 0" ::: "memory");

    // ---- reload P as A-layout fragment (in-wave transpose via LDS) ----
    bf16x16 ap = load_frag(myp + lr * 32, hi);

    // ---- O += P @ V  (B operand = V^T rows, contiguous in t) ----
#pragma unroll
    for (int t4 = 0; t4 < 4; ++t4) {
      const bf16* vrow = Vb + (long)(t4 * 16 + lr) * Tc + kb;
      bf16x16 bv = load_frag(vrow, hi);
      accO[t4] = wmma_bf16(ap, bv, accO[t4]);
    }
  }

  // ---- normalize and store O as [B,T,H,D] ----
  float inv[8];
#pragma unroll
  for (int v = 0; v < 8; ++v) inv[v] = 1.f / lrun[v];
#pragma unroll
  for (int t4 = 0; t4 < 4; ++t4) {
    int d = t4 * 16 + lr;
#pragma unroll
    for (int v = 0; v < 8; ++v) {
      int t = q0 + v + 8 * hi;
      O[(((long)b * Tc + t) * Hc + h) * Dc + d] = (bf16)(accO[t4][v] * inv[v]);
    }
  }
}

// ---------------------------------------------------------------------------
// Output projection: out[m,n] = sum_k Oa[m,k]*Wp[n,k] + bias[n], fp32 out.
// ---------------------------------------------------------------------------
__global__ __launch_bounds__(256)
void gemm_proj(const bf16* __restrict__ Xo, const bf16* __restrict__ W,
               const float* __restrict__ bias, float* __restrict__ out) {
  int tid  = threadIdx.x;
  int lane = tid & 31;
  int lr   = lane & 15;
  int hi   = lane >> 4;
  int wid  = tid >> 5;
  int waveM = wid & 3, waveN = wid >> 2;
  int m0blk = blockIdx.x * 128;
  int n0blk = blockIdx.y * 128;
  int m0 = m0blk + waveM * 32;
  int n0 = n0blk + waveN * 64;

  f32x8 acc[2][4];
  gemm_mainloop(Xo, W, m0blk, n0blk, tid, acc);

#pragma unroll
  for (int j = 0; j < 4; ++j) {
    int gn = n0 + j * 16 + lr;
    float bv = bias[gn];
#pragma unroll
    for (int i = 0; i < 2; ++i) {
#pragma unroll
      for (int v = 0; v < 8; ++v) {
        int m = m0 + i * 16 + v + 8 * hi;
        out[(long)m * Cc + gn] = acc[i][j][v] + bv;
      }
    }
  }
}

// ---------------------------------------------------------------------------
// Launch: cvt(x, Wq, Wk, Wv, Wp) -> 3x gemm_qkv -> attn -> gemm_proj
// Workspace (bf16 elems): xb 8M | W*4 x 1M | Q 8M | K 8M | V^T 8M | O 8M  ~88MB
// ---------------------------------------------------------------------------
extern "C" void kernel_launch(void* const* d_in, const int* in_sizes, int n_in,
                              void* d_out, int out_size, void* d_ws, size_t ws_size,
                              hipStream_t stream) {
  const float* x  = (const float*)d_in[0];
  const float* Wq = (const float*)d_in[1];
  const float* Wk = (const float*)d_in[2];
  const float* Wv = (const float*)d_in[3];
  const float* Wp = (const float*)d_in[4];
  const float* bp = (const float*)d_in[5];
  float* out = (float*)d_out;

  const long nX = (long)Mrows * Cc;      // 8M
  const long nW = (long)Cc * Cc;         // 1M
  bf16* xb  = (bf16*)d_ws;
  bf16* wqb = xb  + nX;
  bf16* wkb = wqb + nW;
  bf16* wvb = wkb + nW;
  bf16* wpb = wvb + nW;
  bf16* Qb  = wpb + nW;
  bf16* Kb  = Qb  + nX;
  bf16* Vtb = Kb  + nX;
  bf16* Ob  = Vtb + nX;

  auto cvt = [&](const float* s, bf16* d, long n) {
    int n4 = (int)(n / 4);
    cvt_f32_bf16<<<(n4 + 255) / 256, 256, 0, stream>>>(s, d, n4);
  };
  cvt(x,  xb,  nX);
  cvt(Wq, wqb, nW);
  cvt(Wk, wkb, nW);
  cvt(Wv, wvb, nW);
  cvt(Wp, wpb, nW);

  dim3 ggrid(Mrows / 128, Cc / 128);     // 64 x 8
  gemm_qkv<<<ggrid, 256, 0, stream>>>(xb, wqb, Qb,  0);
  gemm_qkv<<<ggrid, 256, 0, stream>>>(xb, wkb, Kb,  0);
  gemm_qkv<<<ggrid, 256, 0, stream>>>(xb, wvb, Vtb, 1);

  dim3 agrid(Tc / 128, Hc, Bc);          // 16 x 16 x 4
  attn_fwd<<<agrid, 256, 0, stream>>>(Qb, Kb, Vtb, Ob);

  gemm_proj<<<ggrid, 256, 0, stream>>>(Ob, wpb, bp, out);
}
